// UpsampleEdgeconv_85779086836270
// MI455X (gfx1250) — matrix operand order repaired and checked by reading, not compiled, not run on hardware
//
#include <hip/hip_runtime.h>

typedef __attribute__((ext_vector_type(2)))  float  v2f;
typedef __attribute__((ext_vector_type(8)))  float  v8f;
typedef __attribute__((ext_vector_type(8)))  __bf16 v8bf;
typedef __attribute__((ext_vector_type(16))) __bf16 v16bf;

#define BATCH 8
#define DIM   64
#define NPTS  2048
#define KNN   16
#define C1    256      // 4*Fin
#define C2    128      // 2*Fin == 2*Fout
#define KW1   9
#define J1    8        // K - KW1 + 1
#define KC1   1152     // 128*9
#define COLS1 16384    // NPTS*J1
#define KC2   4096     // 128*32
#define EPS   1e-5f

// ---------- helpers ----------

__device__ __forceinline__ __bf16 f2bf(float f) {
  unsigned u = __builtin_bit_cast(unsigned, f);
  unsigned r = u + 0x7FFFu + ((u >> 16) & 1u);       // round-to-nearest-even
  unsigned short s = (unsigned short)(r >> 16);
  return __builtin_bit_cast(__bf16, s);
}

// e[b, cin(0..127), n, t]: first 64 channels = central point, last 64 = neighbor-central
__device__ __forceinline__ float edge_feat(const float* __restrict__ x,
                                           const int* __restrict__ idx,
                                           int b, int cin, int n, int t) {
  if (cin < DIM) return x[(b * DIM + cin) * NPTS + n];
  const float* xc = x + (b * DIM + (cin - DIM)) * NPTS;
  int nb = idx[(b * NPTS + n) * KNN + t];
  return xc[nb] - xc[n];
}

__device__ __forceinline__ v16bf pack16(v8bf lo, v8bf hi) {
  return __builtin_shufflevector(lo, hi, 0,1,2,3,4,5,6,7,8,9,10,11,12,13,14,15);
}

// ---------- K-prep: fp32 weights -> bf16 ----------
__global__ void kprep(const float* __restrict__ w1, const float* __restrict__ w2,
                      __bf16* __restrict__ w1b, __bf16* __restrict__ w2b) {
  int i = blockIdx.x * 256 + threadIdx.x;
  const int n1 = C1 * KC1;          // 294912
  const int n2 = C2 * KC2;          // 524288
  if (i < n1) w1b[i] = f2bf(w1[i]);
  else if (i < n1 + n2) w2b[i - n1] = f2bf(w2[i - n1]);
}

// ---------- K0: squared norms ----------
__global__ void ksqn(const float* __restrict__ x, float* __restrict__ sqn) {
  int t = blockIdx.x * 256 + threadIdx.x;
  if (t >= BATCH * NPTS) return;
  int b = t >> 11, n = t & (NPTS - 1);
  const float* xb = x + b * DIM * NPTS + n;
  float s = 0.f;
  #pragma unroll
  for (int c = 0; c < DIM; ++c) { float v = xb[c * NPTS]; s += v * v; }
  sqn[t] = s;
}

// ---------- K2: kNN via f32 WMMA (16x16x4), one wave per 16-query tile ----------
__global__ void kknn(const float* __restrict__ x, const float* __restrict__ sqn,
                     int* __restrict__ oidx) {
  __shared__ float sd[32 * 16];
  __shared__ int   si[32 * 16];

  int wid = blockIdx.x;
  int b   = wid >> 7;
  int q0  = (wid & 127) << 4;
  int lane  = threadIdx.x;
  int l16   = lane & 15;
  int khalf = (lane >> 4) << 1;          // 0 or 2

  const float* xb = x + b * DIM * NPTS;
  int qn = q0 + l16;

  // preload B fragments (queries) for all 16 k-steps: B[k][n] = x[b][k][q0+n]
  float bq[32];
  #pragma unroll
  for (int ks = 0; ks < 16; ++ks) {
    bq[2 * ks + 0] = xb[(ks * 4 + khalf + 0) * NPTS + qn];
    bq[2 * ks + 1] = xb[(ks * 4 + khalf + 1) * NPTS + qn];
  }
  float sqq = sqn[b * NPTS + qn];

  // per-lane top-16 (this lane sees candidates c0 + 8*(lane/16) + 0..7 per tile)
  float dl[16]; int il[16];
  #pragma unroll
  for (int i = 0; i < 16; ++i) { dl[i] = 3.4e38f; il[i] = 0x7fffffff; }
  float wmax = 3.4e38f; int wpos = 0;

  for (int c0 = 0; c0 < NPTS; c0 += 16) {
    int cn = c0 + l16;
    const float* xc = xb + cn;
    v8f acc = {};
    #pragma unroll
    for (int ks = 0; ks < 16; ++ks) {
      v2f a, bb;
      a.x = xc[(ks * 4 + khalf + 0) * NPTS];
      a.y = xc[(ks * 4 + khalf + 1) * NPTS];
      bb.x = bq[2 * ks + 0];
      bb.y = bq[2 * ks + 1];
      acc = __builtin_amdgcn_wmma_f32_16x16x4_f32(false, a, false, bb,
                                                  (short)0, acc, false, false);
    }
    // lane holds dot(cand = c0 + vr + 8*(lane>>4), query = q0 + l16)
    int cbase = c0 + ((lane >> 4) << 3);
    #pragma unroll
    for (int vr = 0; vr < 8; ++vr) {
      int c = cbase + vr;
      float dist = sqn[b * NPTS + c] + sqq - 2.f * acc[vr];
      bool take = (c != qn) && (dist < wmax);
      if (take) {
        #pragma unroll
        for (int i = 0; i < 16; ++i)
          if (i == wpos) { dl[i] = dist; il[i] = c; }
        wmax = dl[0]; wpos = 0;
        #pragma unroll
        for (int i = 1; i < 16; ++i)
          if (dl[i] > wmax) { wmax = dl[i]; wpos = i; }
      }
    }
  }

  // merge lane l and lane l+16 (they share the same query)
  #pragma unroll
  for (int i = 0; i < 16; ++i) { sd[lane * 16 + i] = dl[i]; si[lane * 16 + i] = il[i]; }
  __syncthreads();
  if (lane < 16) {
    int q = q0 + lane;
    float* da = sd + lane * 16;        int* ia = si + lane * 16;
    float* db = sd + (lane + 16) * 16; int* ib = si + (lane + 16) * 16;
    for (int o = 0; o < 16; ++o) {
      float bd = 3.4e38f; int bid = 0x7fffffff; int bj = 0;
      for (int j = 0; j < 32; ++j) {
        float d = (j < 16) ? da[j] : db[j - 16];
        int  id = (j < 16) ? ia[j] : ib[j - 16];
        if (d < bd || (d == bd && id < bid)) { bd = d; bid = id; bj = j; }
      }
      oidx[(b * NPTS + q) * KNN + o] = bid;
      if (bj < 16) da[bj] = 3.4e38f; else db[bj - 16] = 3.4e38f;
    }
  }
}

// ---------- K3: conv_inte as GEMM, 8 waves cooperate; B staged once in LDS ----------
// Workgroup covers all 256 couts x 64 columns; each wave owns cout tiles w and w+8.
__global__ void kconv1(const float* __restrict__ x, const int* __restrict__ idx,
                       const __bf16* __restrict__ w1b, const float* __restrict__ b1,
                       float* __restrict__ hpre) {
  __shared__ v8bf BsV[256];                 // [ (col*2 + half)*2 + g ], 4 KB

  int b    = blockIdx.x >> 8;
  int col0 = (blockIdx.x & 255) * 64;

  int tid  = threadIdx.x;
  int lane = tid & 31;
  int w    = tid >> 5;                      // wave 0..7
  int l16  = lane & 15;
  int h    = lane >> 4;

  int rowA0 = w * 16 + l16;                 // cout tile w
  int rowA1 = (w + 8) * 16 + l16;           // cout tile w+8

  // staging assignment: thread -> (col, half, e0..e0+7)
  int scol = tid >> 2;
  int q    = tid & 3;
  int shh  = q >> 1;
  int sg   = q & 1;
  int scoln = col0 + scol;
  int sn = scoln >> 3, sj = scoln & 7;

  v8f acc[8] = {{}, {}, {}, {}, {}, {}, {}, {}};   // [half4*4 + t4]

  for (int ks = 0; ks < 36; ++ks) {
    // compute this thread's 8 B elements (k = ks*32 + 16*half + 8*g + e)
    v8bf pack;
    #pragma unroll
    for (int e = 0; e < 8; ++e) {
      int k = ks * 32 + 16 * shh + 8 * sg + e;
      unsigned cin = (unsigned)k / 9u;
      int kw = k - (int)cin * 9;
      pack[e] = f2bf(edge_feat(x, idx, b, (int)cin, sn, sj + kw));
    }
    __syncthreads();                        // previous iteration's reads done
    BsV[(scol * 2 + shh) * 2 + sg] = pack;
    __syncthreads();                        // staging visible

    const v8bf* ap0 = (const v8bf*)(w1b + rowA0 * KC1 + ks * 32 + 8 * h);
    v16bf a0 = pack16(ap0[0], ap0[2]);
    const v8bf* ap1 = (const v8bf*)(w1b + rowA1 * KC1 + ks * 32 + 8 * h);
    v16bf a1 = pack16(ap1[0], ap1[2]);

    #pragma unroll
    for (int t4 = 0; t4 < 4; ++t4) {
      int col = t4 * 16 + l16;
      const v8bf* bp = BsV + col * 4 + h;
      v16bf bfrag = pack16(bp[0], bp[2]);
      acc[t4]     = __builtin_amdgcn_wmma_f32_16x16x32_bf16(false, a0, false, bfrag,
                                                            (short)0, acc[t4], false, false);
      acc[4 + t4] = __builtin_amdgcn_wmma_f32_16x16x32_bf16(false, a1, false, bfrag,
                                                            (short)0, acc[4 + t4], false, false);
    }
  }

  #pragma unroll
  for (int half = 0; half < 2; ++half) {
    int m0 = (w + 8 * half) * 16;
    #pragma unroll
    for (int r = 0; r < 8; ++r) {
      int cout = m0 + r + 8 * h;
      float bias = b1[cout];
      #pragma unroll
      for (int t4 = 0; t4 < 4; ++t4) {
        int col = col0 + t4 * 16 + l16;
        hpre[(size_t)(b * C1 + cout) * COLS1 + col] = acc[half * 4 + t4][r] + bias;
      }
    }
  }
}

// ---------- BN stats (deterministic two-level reduction) ----------
__global__ void kstats1(const float* __restrict__ hpre, const float* __restrict__ g,
                        const float* __restrict__ be, float* __restrict__ scale,
                        float* __restrict__ shift) {
  int c = blockIdx.x, tid = threadIdx.x;
  float s = 0.f, s2 = 0.f;
  for (int i = tid; i < BATCH * COLS1; i += 256) {
    int bb = i >> 14, col = i & (COLS1 - 1);
    float v = hpre[(size_t)(bb * C1 + c) * COLS1 + col];
    s += v; s2 += v * v;
  }
  __shared__ float a1[256], a2[256];
  a1[tid] = s; a2[tid] = s2; __syncthreads();
  for (int off = 128; off; off >>= 1) {
    if (tid < off) { a1[tid] += a1[tid + off]; a2[tid] += a2[tid + off]; }
    __syncthreads();
  }
  if (tid == 0) {
    const float M = (float)(BATCH * COLS1);
    float m = a1[0] / M;
    float var = a2[0] / M - m * m;
    float rs = rsqrtf(var + EPS);
    scale[c] = g[c] * rs;
    shift[c] = be[c] - m * rs * g[c];
  }
}

__global__ void kstats2(const float* __restrict__ ypre, const float* __restrict__ g,
                        const float* __restrict__ be, float* __restrict__ scale,
                        float* __restrict__ shift) {
  int c = blockIdx.x, tid = threadIdx.x;
  float s = 0.f, s2 = 0.f;
  for (int i = tid; i < BATCH * NPTS; i += 256) {
    int bb = i >> 11, n = i & (NPTS - 1);
    float v = ypre[(size_t)(bb * C2 + c) * NPTS + n];
    s += v; s2 += v * v;
  }
  __shared__ float a1[256], a2[256];
  a1[tid] = s; a2[tid] = s2; __syncthreads();
  for (int off = 128; off; off >>= 1) {
    if (tid < off) { a1[tid] += a1[tid + off]; a2[tid] += a2[tid + off]; }
    __syncthreads();
  }
  if (tid == 0) {
    const float M = (float)(BATCH * NPTS);
    float m = a1[0] / M;
    float var = a2[0] / M - m * m;
    float rs = rsqrtf(var + EPS);
    scale[c] = g[c] * rs;
    shift[c] = be[c] - m * rs * g[c];
  }
}

// ---------- K4: conv2 as GEMM, 8 waves cooperate; B staged once in LDS ----------
// merge[b,c,n,t]: t<16 -> edge feature, t>=16 -> bn1+leakyrelu(h_pre[b,2c+s,n,p])
__global__ void kconv2(const float* __restrict__ x, const int* __restrict__ idx,
                       const __bf16* __restrict__ w2b, const float* __restrict__ b2,
                       const float* __restrict__ hpre, const float* __restrict__ scale1,
                       const float* __restrict__ shift1, float* __restrict__ ypre) {
  __shared__ v8bf BsV[256];                 // 4 KB
  __shared__ float sc1[C1], sh1[C1];

  int tid  = threadIdx.x;
  if (tid < C1) { sc1[tid] = scale1[tid]; sh1[tid] = shift1[tid]; }
  __syncthreads();

  int b    = blockIdx.x >> 5;
  int col0 = (blockIdx.x & 31) * 64;

  int lane = tid & 31;
  int w    = tid >> 5;                      // wave 0..7 owns cout tile w
  int l16  = lane & 15;
  int h    = lane >> 4;
  int rowA = w * 16 + l16;

  int scol = tid >> 2;
  int q    = tid & 3;
  int shh  = q >> 1;
  int sg   = q & 1;
  int sn   = col0 + scol;                   // column == point index n

  v8f acc[4] = {{}, {}, {}, {}};

  for (int ks = 0; ks < 128; ++ks) {        // ks == input channel c of merge
    v8bf pack;
    if (shh == 0) {                         // t = 8*sg + e in 0..15: edge features
      #pragma unroll
      for (int e = 0; e < 8; ++e)
        pack[e] = f2bf(edge_feat(x, idx, b, ks, sn, 8 * sg + e));
    } else {                                // t = 16 + 8*sg + e: shuffled conv1 output
      int c1 = 2 * ks + sg;                 // s = sg, p = e
      float sc = sc1[c1], sh = sh1[c1];
      const float* hp = hpre + (size_t)(b * C1 + c1) * COLS1 + sn * 8;
      #pragma unroll
      for (int e = 0; e < 8; ++e) {
        float hv = hp[e] * sc + sh;
        pack[e] = f2bf(hv > 0.f ? hv : 0.2f * hv);
      }
    }
    __syncthreads();
    BsV[(scol * 2 + shh) * 2 + sg] = pack;
    __syncthreads();

    const v8bf* ap = (const v8bf*)(w2b + rowA * KC2 + ks * 32 + 8 * h);
    v16bf afrag = pack16(ap[0], ap[2]);
    #pragma unroll
    for (int t4 = 0; t4 < 4; ++t4) {
      int col = t4 * 16 + l16;
      const v8bf* bp = BsV + col * 4 + h;
      v16bf bfrag = pack16(bp[0], bp[2]);
      acc[t4] = __builtin_amdgcn_wmma_f32_16x16x32_bf16(false, afrag, false, bfrag,
                                                        (short)0, acc[t4], false, false);
    }
  }

  #pragma unroll
  for (int r = 0; r < 8; ++r) {
    int cout = w * 16 + r + 8 * h;
    float bias = b2[cout];
    #pragma unroll
    for (int t4 = 0; t4 < 4; ++t4) {
      int n = col0 + t4 * 16 + l16;
      ypre[(size_t)(b * C2 + cout) * NPTS + n] = acc[t4][r] + bias;
    }
  }
}

// ---------- K5: bn2 + relu + reshape [B,128,N] -> [B,64,2N] ----------
__global__ void kfinal(const float* __restrict__ ypre, const float* __restrict__ scale2,
                       const float* __restrict__ shift2, float* __restrict__ out) {
  int i = blockIdx.x * 256 + threadIdx.x;
  if (i >= BATCH * C2 * NPTS) return;
  int n = i & (NPTS - 1);
  int c = (i >> 11) & (C2 - 1);
  int bb = i >> 18;
  float v = ypre[i] * scale2[c] + shift2[c];
  v = v > 0.f ? v : 0.f;
  int f = c >> 1, s = c & 1;
  out[((size_t)(bb * 64 + f) << 12) + (s << 11) + n] = v;
}

// ---------- launcher ----------
extern "C" void kernel_launch(void* const* d_in, const int* in_sizes, int n_in,
                              void* d_out, int out_size, void* d_ws, size_t ws_size,
                              hipStream_t stream) {
  const float* x    = (const float*)d_in[0];
  const float* w1   = (const float*)d_in[1];
  const float* b1   = (const float*)d_in[2];
  const float* g1   = (const float*)d_in[3];
  const float* be1  = (const float*)d_in[4];
  const float* w2   = (const float*)d_in[5];
  const float* b2   = (const float*)d_in[6];
  const float* g2   = (const float*)d_in[7];
  const float* be2  = (const float*)d_in[8];
  float* out = (float*)d_out;

  char* ws = (char*)d_ws;
  float*  sqn    = (float*)(ws + 0);                 //  65536 B
  int*    idx    = (int*)  (ws + 65536);             //  1 MB
  float*  scale1 = (float*)(ws + 1114112);
  float*  shift1 = (float*)(ws + 1115136);
  float*  scale2 = (float*)(ws + 1116160);
  float*  shift2 = (float*)(ws + 1116672);
  __bf16* w1b    = (__bf16*)(ws + 1117184);          //  576 KB
  __bf16* w2b    = (__bf16*)(ws + 1707008);          //  1 MB
  float*  ypre   = (float*)(ws + 2755584);           //  8 MB
  float*  hpre   = (float*)(ws + 11144192);          //  128 MB

  kprep  <<<(C1*KC1 + C2*KC2 + 255) / 256, 256, 0, stream>>>(w1, w2, w1b, w2b);
  ksqn   <<<(BATCH * NPTS) / 256, 256, 0, stream>>>(x, sqn);
  kknn   <<<BATCH * (NPTS / 16), 32, 0, stream>>>(x, sqn, idx);
  kconv1 <<<BATCH * 256, 256, 0, stream>>>(x, idx, w1b, b1, hpre);
  kstats1<<<C1, 256, 0, stream>>>(hpre, g1, be1, scale1, shift1);
  kconv2 <<<BATCH * 32, 256, 0, stream>>>(x, idx, w2b, b2, hpre, scale1, shift1, ypre);
  kstats2<<<C2, 256, 0, stream>>>(ypre, g2, be2, scale2, shift2);
  kfinal <<<(BATCH * C2 * NPTS) / 256, 256, 0, stream>>>(ypre, scale2, shift2, out);
}
